// TheoremGenerator_953482739987
// MI455X (gfx1250) — matrix operand order repaired and checked by reading, not compiled, not run on hardware
//
#include <hip/hip_runtime.h>
#include <hip/hip_bf16.h>
#include <math.h>

typedef __bf16 bf16_t;
typedef bf16_t bf16x16 __attribute__((ext_vector_type(16)));
typedef float  f32x8   __attribute__((ext_vector_type(8)));

// ---------------------------------------------------------------- utilities
__device__ __forceinline__ unsigned wang_hash(unsigned s) {
  s = (s ^ 61u) ^ (s >> 16); s *= 9u; s ^= s >> 4; s *= 0x27d4eb2du; s ^= s >> 15;
  return s;
}

// ------------------------------------------------------------ fp32 -> bf16
__global__ __launch_bounds__(256) void f2bf_kernel(const float* __restrict__ src,
                                                   bf16_t* __restrict__ dst, int n) {
  int i = blockIdx.x * 256 + threadIdx.x;
  if (i < n) dst[i] = (bf16_t)src[i];
}

// ---------------------------------------------------- pack B into WMMA tiles
// Packed layout: [ntile][ktile][lane(32)][dword(8)] ; lane = (n&15) | ((k16half)<<4)
// dword d holds bf16 pair (k = kt*32 + half*16 + 2d, 2d+1).
// B[k][n] = (k<K0 ? src0 : src1), optionally transposed (B[k][n] = S[n][k]).
__global__ __launch_bounds__(256) void pack_b_kernel(
    const float* __restrict__ src0, const float* __restrict__ src1,
    unsigned int* __restrict__ dst, int K, int N, int K0, int ldsrc, int transpose) {
  size_t o = (size_t)blockIdx.x * 256 + threadIdx.x;
  size_t blk = (size_t)8 * (size_t)K;            // dwords per 16-column ntile
  int ntile = (int)(o / blk);
  size_t rem = o % blk;
  int ktile = (int)(rem >> 8);
  int r2 = (int)(rem & 255);
  int lane = r2 >> 3, d = r2 & 7;
  int n = ntile * 16 + (lane & 15);
  int kb = ktile * 32 + ((lane >> 4) << 4) + d * 2;
  float v[2];
#pragma unroll
  for (int e = 0; e < 2; ++e) {
    int k = kb + e;
    const float* s = src0; int kk = k;
    if (k >= K0) { s = src1; kk = k - K0; }
    v[e] = transpose ? s[(size_t)n * ldsrc + kk] : s[(size_t)kk * N + n];
  }
  union { bf16_t h[2]; unsigned int u; } p;
  p.h[0] = (bf16_t)v[0]; p.h[1] = (bf16_t)v[1];
  dst[o] = p.u;
}

// ------------------------------------------------------- WMMA GEMM (M = 128)
// C[128 x N] = A[128 x K](bf16) * Bpacked + bias0 + bias1, scaled by 1/temp.
// Block: 256 threads = 8 waves; each wave owns a 16-row strip; block owns 128
// columns (8 WMMA D-tiles per wave).  The A K-slab (128x32 bf16, 8 KB) is
// double-buffered in LDS: slab kt+1 is async-copied while slab kt is consumed.
#define GEMM_NT 8
__global__ __launch_bounds__(256) void gemm_wmma_kernel(
    const bf16_t* __restrict__ A, const unsigned int* __restrict__ Bp,
    float* __restrict__ C, long long ldc,
    const float* __restrict__ bias0, const float* __restrict__ bias1,
    const int* __restrict__ tempPtr, int N, int K) {
  __shared__ __align__(16) unsigned char Asm[2][128 * 64];  // 2 x (128 rows x 32 bf16)
  const int tid = threadIdx.x;
  const int lane = tid & 31;
  const int w = tid >> 5;
  const int nbase = blockIdx.x * (GEMM_NT * 16);
  const int ktiles = K >> 5;

  f32x8 acc[GEMM_NT];
#pragma unroll
  for (int j = 0; j < GEMM_NT; ++j)
    acc[j] = (f32x8){0.f, 0.f, 0.f, 0.f, 0.f, 0.f, 0.f, 0.f};

  const int rowA = (w << 4) + (lane & 15);
  const int hi16 = lane >> 4;

  auto issue_slab = [&](int kt, int buf) {
    const int kc = kt << 5;
#pragma unroll
    for (int s = 0; s < 2; ++s) {
      int i = tid * 2 + s;                       // 0..511 sixteen-byte chunks
      int r = i >> 2;
      int c4 = i & 3;
      const bf16_t* src = A + (size_t)r * K + kc + c4 * 8;
      unsigned int la = (unsigned int)(unsigned long long)(const void*)(&Asm[buf][i * 16]);
      unsigned long long ga = (unsigned long long)(const void*)src;
      asm volatile("global_load_async_to_lds_b128 %0, %1, off"
                   :: "v"(la), "v"(ga) : "memory");
    }
  };

  issue_slab(0, 0);

  for (int kt = 0; kt < ktiles; ++kt) {
    asm volatile("s_wait_asynccnt 0" ::: "memory");  // slab kt landed (this wave)
    __syncthreads();                                 // all waves: slab kt visible,
                                                     // slab kt-1 buffer free
    if (kt + 1 < ktiles) issue_slab(kt + 1, (kt + 1) & 1);

    // A fragment (16x32 bf16, contiguous k = 16*half + s, matching packed B) from current buffer
    union { unsigned int u[8]; bf16x16 v; } af;
    {
      const unsigned char* base = &Asm[kt & 1][rowA * 64 + (hi16 ? 32 : 0)];
      uint4 lo = *(const uint4*)(base);
      uint4 hq = *(const uint4*)(base + 16);
      af.u[0] = lo.x; af.u[1] = lo.y; af.u[2] = lo.z; af.u[3] = lo.w;
      af.u[4] = hq.x; af.u[5] = hq.y; af.u[6] = hq.z; af.u[7] = hq.w;
    }

#pragma unroll
    for (int j = 0; j < GEMM_NT; ++j) {
      const int nt = (nbase >> 4) + j;
      const unsigned int* bptr = Bp + (((size_t)nt * ktiles + kt) << 8) + lane * 8;
      __builtin_prefetch(bptr + 256, 0, 0);          // next K-slab of this tile
      union { unsigned int u[8]; bf16x16 v; } bfr;
      uint4 b0 = *(const uint4*)(bptr);
      uint4 b1 = *(const uint4*)(bptr + 4);
      bfr.u[0] = b0.x; bfr.u[1] = b0.y; bfr.u[2] = b0.z; bfr.u[3] = b0.w;
      bfr.u[4] = b1.x; bfr.u[5] = b1.y; bfr.u[6] = b1.z; bfr.u[7] = b1.w;
      acc[j] = __builtin_amdgcn_wmma_f32_16x16x32_bf16(
          false, af.v, false, bfr.v, (short)0, acc[j], false, false);
    }
  }

  float scale = 1.0f;
  if (tempPtr) { int tv = *tempPtr; scale = 1.0f / (float)(tv == 0 ? 1 : tv); }
  const int rowD = (w << 4) + (hi16 ? 8 : 0);
  const int colD = nbase + (lane & 15);
#pragma unroll
  for (int j = 0; j < GEMM_NT; ++j) {
    int col = colD + j * 16;
    float bv = 0.f;
    if (bias0) bv += bias0[col];
    if (bias1) bv += bias1[col];
#pragma unroll
    for (int r = 0; r < 8; ++r)
      C[(size_t)(rowD + r) * (size_t)ldc + col] = (acc[j][r] + bv) * scale;
  }
}

// ----------------------------------------------------- LayerNorm + ReLU row
__global__ __launch_bounds__(256) void ln_relu_kernel(
    const float* __restrict__ X, const float* __restrict__ gam,
    const float* __restrict__ bet, int N,
    bf16_t* dst0, int ld0, int off0, bf16_t* dst1, int ld1, int off1) {
  __shared__ float s1[256], s2[256];
  int b = blockIdx.x, tid = threadIdx.x;
  const float* x = X + (size_t)b * N;
  float s = 0.f, q = 0.f;
  for (int j = tid; j < N; j += 256) { float v = x[j]; s += v; q += v * v; }
  s1[tid] = s; s2[tid] = q; __syncthreads();
  for (int st = 128; st > 0; st >>= 1) {
    if (tid < st) { s1[tid] += s1[tid + st]; s2[tid] += s2[tid + st]; }
    __syncthreads();
  }
  float mean = s1[0] / N;
  float var = s2[0] / N - mean * mean;
  float inv = rsqrtf(var + 1e-5f);
  for (int j = tid; j < N; j += 256) {
    float y = (x[j] - mean) * inv * gam[j] + bet[j];
    y = y > 0.f ? y : 0.f;
    dst0[(size_t)b * ld0 + off0 + j] = (bf16_t)y;
    if (dst1) dst1[(size_t)b * ld1 + off1 + j] = (bf16_t)y;
  }
}

// ------------------------------------------------------- LSTM cell pointwise
__global__ __launch_bounds__(256) void lstm_elem_kernel(
    const float* __restrict__ gates, float* __restrict__ cst,
    bf16_t* dA, int ldA, int offA, bf16_t* dB, int ldB, int offB) {
  int idx = blockIdx.x * 256 + threadIdx.x;     // b*512 + j
  int b = idx >> 9, j = idx & 511;
  const float* g = gates + (size_t)b * 2048;
  float gi = g[j], gf = g[512 + j], gg = g[1024 + j], go = g[1536 + j];
  float si = 1.f / (1.f + __expf(-gi));
  float sf = 1.f / (1.f + __expf(-gf));
  float so = 1.f / (1.f + __expf(-go));
  float c = sf * cst[idx] + si * tanhf(gg);
  cst[idx] = c;
  float h = so * tanhf(c);
  dA[(size_t)b * ldA + offA + j] = (bf16_t)h;
  dB[(size_t)b * ldB + offB + j] = (bf16_t)h;
}

// ------------------------------------------------- init: cells, start tokens
__global__ __launch_bounds__(256) void init_state_kernel(
    float* c0, float* c1, int* tokens, const float* __restrict__ tok_table,
    bf16_t* A0) {
  int idx = blockIdx.x * 256 + threadIdx.x;     // b*512 + j
  int b = idx >> 9, j = idx & 511;
  c0[idx] = 0.f; c1[idx] = 0.f;
  A0[(size_t)b * 1024 + j] = (bf16_t)tok_table[j];  // token 0 embedding row
  if (idx < 128) tokens[idx * 31] = 0;
}

// --------------------------------- Gumbel-max sample + embedding gather
__global__ __launch_bounds__(256) void sample_gather_kernel(
    const float* __restrict__ logits, int t, int* tokens,
    const float* __restrict__ tok_table, bf16_t* A0) {
  __shared__ float smax[256];
  __shared__ int sidx[256];
  int b = blockIdx.x, tid = threadIdx.x;
  float best = -1e30f; int bi = 0;
  for (int v = tid; v < 32000; v += 256) {
    float x = logits[(size_t)b * 960000 + v];
    unsigned h = wang_hash((unsigned)(b * 32000 + v) ^ ((unsigned)t * 0x9E3779B9u));
    float u = (float)(h >> 8) * (1.0f / 16777216.0f) + 1e-7f;
    float gum = -__logf(-__logf(u));
    float val = x + gum;
    if (val > best) { best = val; bi = v; }
  }
  smax[tid] = best; sidx[tid] = bi; __syncthreads();
  for (int st = 128; st > 0; st >>= 1) {
    if (tid < st && smax[tid + st] > smax[tid]) {
      smax[tid] = smax[tid + st]; sidx[tid] = sidx[tid + st];
    }
    __syncthreads();
  }
  int tok = sidx[0];
  if (tid == 0) tokens[b * 31 + t + 1] = tok;
  for (int j = tid; j < 512; j += 256)
    A0[(size_t)b * 1024 + j] = (bf16_t)tok_table[(size_t)tok * 512 + j];
}

// ============================================================= host driver
extern "C" void kernel_launch(void* const* d_in, const int* in_sizes, int n_in,
                              void* d_out, int out_size, void* d_ws, size_t ws_size,
                              hipStream_t stream) {
  (void)in_sizes; (void)n_in; (void)out_size; (void)ws_size;
  const float* SE   = (const float*)d_in[0];
  const float* Wa   = (const float*)d_in[1];
  const float* ba   = (const float*)d_in[2];
  const float* g1   = (const float*)d_in[3];
  const float* b1   = (const float*)d_in[4];
  const float* Wp   = (const float*)d_in[5];
  const float* bp   = (const float*)d_in[6];
  const float* g2   = (const float*)d_in[7];
  const float* b2   = (const float*)d_in[8];
  const float* tok  = (const float*)d_in[9];
  const float* Wih0 = (const float*)d_in[10];
  const float* Whh0 = (const float*)d_in[11];
  const float* bih0 = (const float*)d_in[12];
  const float* bhh0 = (const float*)d_in[13];
  const float* Wih1 = (const float*)d_in[14];
  const float* Whh1 = (const float*)d_in[15];
  const float* bih1 = (const float*)d_in[16];
  const float* bhh1 = (const float*)d_in[17];
  const float* Wout = (const float*)d_in[18];
  const float* bout = (const float*)d_in[19];
  const int*   temp = (const int*)d_in[20];

  int*   tokens = (int*)d_out;                     // [128, 31]
  float* logits = (float*)d_out + 128 * 31;        // [128, 30, 32000]

  char* ws = (char*)d_ws;
  auto alloc = [&](size_t bytes) -> char* {
    char* p = ws; ws += (bytes + 255) & ~(size_t)255; return p;
  };
  unsigned int* WoutP = (unsigned int*)alloc((size_t)512 * 32000 * 2);
  unsigned int* Wc0P  = (unsigned int*)alloc((size_t)1024 * 2048 * 2);
  unsigned int* Wc1P  = (unsigned int*)alloc((size_t)1024 * 2048 * 2);
  unsigned int* WaP   = (unsigned int*)alloc((size_t)1024 * 1024 * 2);
  unsigned int* WpP   = (unsigned int*)alloc((size_t)1024 * 512 * 2);
  bf16_t* Ase  = (bf16_t*)alloc((size_t)128 * 1024 * 2);
  bf16_t* Apre = (bf16_t*)alloc((size_t)128 * 1024 * 2);
  bf16_t* A0   = (bf16_t*)alloc((size_t)128 * 1024 * 2);  // [emb | h0]
  bf16_t* A1   = (bf16_t*)alloc((size_t)128 * 1024 * 2);  // [h0  | h1]
  bf16_t* h1b  = (bf16_t*)alloc((size_t)128 * 512 * 2);
  float* tmp1  = (float*)alloc((size_t)128 * 1024 * 4);
  float* tmp2  = (float*)alloc((size_t)128 * 512 * 4);
  float* gates = (float*)alloc((size_t)128 * 2048 * 4);
  float* c0    = (float*)alloc((size_t)128 * 512 * 4);
  float* c1    = (float*)alloc((size_t)128 * 512 * 4);

  // --- weight packing (fp32 -> bf16 WMMA-B tiles), recomputed each call ---
  pack_b_kernel<<<2048, 256, 0, stream>>>(Wa, Wa, WaP, 1024, 1024, 1024, 0, 0);
  pack_b_kernel<<<1024, 256, 0, stream>>>(Wp, Wp, WpP, 1024, 512, 1024, 0, 0);
  pack_b_kernel<<<4096, 256, 0, stream>>>(Wih0, Whh0, Wc0P, 1024, 2048, 512, 512, 1);
  pack_b_kernel<<<4096, 256, 0, stream>>>(Wih1, Whh1, Wc1P, 1024, 2048, 512, 512, 1);
  pack_b_kernel<<<32000, 256, 0, stream>>>(Wout, Wout, WoutP, 512, 32000, 512, 0, 0);

  // --- preprocessing: adapter + proj ---
  f2bf_kernel<<<512, 256, 0, stream>>>(SE, Ase, 128 * 1024);
  gemm_wmma_kernel<<<8, 256, 0, stream>>>(Ase, WaP, tmp1, 1024, ba, nullptr, nullptr, 1024, 1024);
  ln_relu_kernel<<<128, 256, 0, stream>>>(tmp1, g1, b1, 1024, Apre, 1024, 0, nullptr, 0, 0);
  gemm_wmma_kernel<<<4, 256, 0, stream>>>(Apre, WpP, tmp2, 512, bp, nullptr, nullptr, 512, 1024);
  // h -> initial hidden of both LSTM layers (bf16 halves of A0/A1)
  ln_relu_kernel<<<128, 256, 0, stream>>>(tmp2, g2, b2, 512, A0, 1024, 512, A1, 1024, 512);
  init_state_kernel<<<256, 256, 0, stream>>>(c0, c1, tokens, tok, A0);

  // --- 30 decode steps ---
  for (int t = 0; t < 30; ++t) {
    gemm_wmma_kernel<<<16, 256, 0, stream>>>(A0, Wc0P, gates, 2048, bih0, bhh0, nullptr, 2048, 1024);
    lstm_elem_kernel<<<256, 256, 0, stream>>>(gates, c0, A1, 1024, 0, A0, 1024, 512);
    gemm_wmma_kernel<<<16, 256, 0, stream>>>(A1, Wc1P, gates, 2048, bih1, bhh1, nullptr, 2048, 1024);
    lstm_elem_kernel<<<256, 256, 0, stream>>>(gates, c1, A1, 1024, 512, h1b, 512, 0);
    gemm_wmma_kernel<<<250, 256, 0, stream>>>(h1b, WoutP, logits + (size_t)t * 32000,
                                              960000, bout, nullptr, temp, 32000, 512);
    sample_gather_kernel<<<128, 256, 0, stream>>>(logits + (size_t)t * 32000, t,
                                                  tokens, tok, A0);
  }
}
